// Attention_62818191671440
// MI455X (gfx1250) — matrix operand order repaired and checked by reading, not compiled, not run on hardware
//
#include <hip/hip_runtime.h>
#include <hip/hip_bf16.h>

// ---------------------------------------------------------------------------
// CDNA5 (gfx1250, wave32) attention block:
//   q/k/v = W{q,k,v} @ x ; per-head flash attention ; out = Wo @ attn + bo
// All matmuls run on v_wmma_f32_16x16x32_f16 (f16 in, f32 accumulate).
// ~34 GFLOP total, ~33 MB HBM traffic -> compute-bound; scores (268 MB if
// materialized, > 192 MB L2) are streamed flash-style and never hit memory.
// __launch_bounds__(256,1): allow full VGPR budget (no scratch spills in the
// flash inner loop; latency is hidden by the wmma/ds dependency chains).
// ---------------------------------------------------------------------------

typedef __attribute__((ext_vector_type(16))) _Float16 v16h;
typedef __attribute__((ext_vector_type(8)))  float    v8f;

#define HEADS 16
#define DHEAD 64
#define CDIM  1024   // QUERY_DIM == INNER
#define SEQ   2048

__device__ __forceinline__ v8f zero8() {
  v8f z = {0.f,0.f,0.f,0.f,0.f,0.f,0.f,0.f};
  return z;
}

// A-fragment (16x32 f16, M x K). ISA layout: lanes 0-15 hold M=0..15,
// lanes 16-31 repeat M with K offset +8; elements 0..7 cover K base {0..7},
// elements 8..15 cover K base {16..23}.  ld = row stride in halfs.
__device__ __forceinline__ v16h frag_a(const _Float16* p, int ld) {
  const int lane = threadIdx.x & 31;
  const int m  = lane & 15;
  const int hi = lane >> 4;
  const _Float16* row = p + m * ld + hi * 8;
  v16h f;
#pragma unroll
  for (int j = 0; j < 4; ++j) { f[2*j]   = row[2*j];      f[2*j+1] = row[2*j+1]; }
#pragma unroll
  for (int j = 0; j < 4; ++j) { f[8+2*j] = row[16+2*j];   f[9+2*j] = row[16+2*j+1]; }
  return f;
}

// B-fragment (32x16 f16, K x N). ISA layout: lane l holds column N=l&15;
// elements 0..7 are K=8*(l>>4)+0..7, elements 8..15 are K=16+8*(l>>4)+0..7.
__device__ __forceinline__ v16h frag_b(const _Float16* p, int ld) {
  const int lane = threadIdx.x & 31;
  const int n  = lane & 15;
  const int hi = lane >> 4;
  v16h f;
#pragma unroll
  for (int j = 0; j < 8; ++j) { f[j] = p[(hi * 8 + j) * ld + n]; f[8 + j] = p[(16 + hi * 8 + j) * ld + n]; }
  return f;
}

__device__ __forceinline__ v8f wmma32(v16h a, v16h b, v8f c) {
  // D = A x B + C  (16x16x32 f16 -> f32), no neg/reuse modifiers
  return __builtin_amdgcn_wmma_f32_16x16x32_f16(
      false, a, false, b, (short)0, c, false, false);
}

// ---------------------------------------------------------------------------
// Kernel 1: fp32 -> f16 conversion (x + weights into workspace)
// ---------------------------------------------------------------------------
__global__ void __launch_bounds__(256)
cvt_f32_f16(const float* __restrict__ src, _Float16* __restrict__ dst, int n) {
  int i = blockIdx.x * blockDim.x + threadIdx.x;
  if (i < n) dst[i] = (_Float16)src[i];
}

// ---------------------------------------------------------------------------
// Kernel 2/4: tiled WMMA GEMM  C[M][N] = A[M][K] * B[K][N]  (+bias, fp32 out)
// Workgroup = 256 threads (8 waves), tile 128x128, K-step 32.
// Wave (wm 0..3, wn 0..1) owns a 32x64 sub-tile = 2x4 wmma accumulators.
// LDS row padding (A ld=40, B ld=136 halfs) keeps fragment reads conflict-free.
// ---------------------------------------------------------------------------
template<bool F32OUT>
__global__ void __launch_bounds__(256, 1)
gemm_wmma(const _Float16* __restrict__ A,
          const _Float16* __restrict__ B,
          _Float16* __restrict__ Ch,
          float*    __restrict__ Cf,
          const float* __restrict__ bias,
          int M, int N, int K) {
  __shared__ _Float16 As[128][40];
  __shared__ _Float16 Bs[32][136];

  const int tid  = threadIdx.x;
  const int wid  = tid >> 5;
  const int wm   = wid >> 1;          // 0..3 (M)
  const int wn   = wid & 1;           // 0..1 (N)
  const int row0 = blockIdx.y * 128;
  const int col0 = blockIdx.x * 128;

  v8f acc[2][4];
#pragma unroll
  for (int i = 0; i < 2; ++i)
#pragma unroll
    for (int j = 0; j < 4; ++j) acc[i][j] = zero8();

  for (int k0 = 0; k0 < K; k0 += 32) {
    // cooperative tile loads, 16B per transaction
    for (int c = tid; c < 512; c += 256) {            // A: 128 rows x 4 chunks
      int r = c >> 2, kc = (c & 3) * 8;
      *(uint4*)(&As[r][kc]) =
          *(const uint4*)(A + (size_t)(row0 + r) * K + k0 + kc);
    }
    for (int c = tid; c < 512; c += 256) {            // B: 32 rows x 16 chunks
      int r = c >> 4, nc = (c & 15) * 8;
      *(uint4*)(&Bs[r][nc]) =
          *(const uint4*)(B + (size_t)(k0 + r) * N + col0 + nc);
    }
    if (k0 + 32 < K) {                                // prefetch next K tile
      __builtin_prefetch(A + (size_t)(row0 + (tid >> 1)) * K + k0 + 32, 0, 1);
      __builtin_prefetch(B + (size_t)(k0 + 32 + (tid & 31)) * N + col0, 0, 1);
    }
    __syncthreads();

    v16h a0 = frag_a(&As[wm * 32][0],      40);
    v16h a1 = frag_a(&As[wm * 32 + 16][0], 40);
#pragma unroll
    for (int nt = 0; nt < 4; ++nt) {
      v16h b = frag_b(&Bs[0][wn * 64 + nt * 16], 136);
      acc[0][nt] = wmma32(a0, b, acc[0][nt]);
      acc[1][nt] = wmma32(a1, b, acc[1][nt]);
    }
    __syncthreads();
  }

  // C/D layout: lane -> n = lane&15, element r -> m = r + 8*(lane>>4)
  const int lane = tid & 31;
  const int nl = lane & 15, hi = lane >> 4;
#pragma unroll
  for (int mt = 0; mt < 2; ++mt)
#pragma unroll
    for (int nt = 0; nt < 4; ++nt) {
      int r0 = row0 + wm * 32 + mt * 16;
      int c0 = col0 + wn * 64 + nt * 16 + nl;
#pragma unroll
      for (int r = 0; r < 8; ++r) {
        int m = r0 + r + hi * 8;
        if (F32OUT) {
          float bv = bias ? bias[m] : 0.f;
          Cf[(size_t)m * N + c0] = acc[mt][nt][r] + bv;
        } else {
          Ch[(size_t)m * N + c0] = (_Float16)acc[mt][nt][r];
        }
      }
    }
}

// ---------------------------------------------------------------------------
// Kernel 3: flash attention, one WG per (head, 128-query block).
// 8 waves, each owns 16 queries; Q fragments live in registers; K/V stream
// through LDS in 64-key blocks; online softmax; P goes through per-wave LDS
// to reach A-fragment layout for the PV wmma.
// q/k/v layout: [o = h*64+d][s] f16.
// ---------------------------------------------------------------------------
__global__ void __launch_bounds__(256, 1)
attn_flash_wmma(const _Float16* __restrict__ Q,
                const _Float16* __restrict__ Km,
                const _Float16* __restrict__ V,
                _Float16* __restrict__ O) {
  __shared__ _Float16 Qs[128][72];       // [sq][d]   18.0 KB
  __shared__ _Float16 Ks[64][72];        // [d][sk]    9.0 KB
  __shared__ _Float16 Vs[64][72];        // [sk][d]    9.0 KB
  __shared__ _Float16 Ps[8][16][72];     // per-wave P 18.0 KB  (54 KB total)

  const int tid  = threadIdx.x;
  const int wid  = tid >> 5;
  const int lane = tid & 31;
  const int nl = lane & 15, hi = lane >> 4;
  const int h  = blockIdx.y;
  const int q0 = blockIdx.x * 128;
  const size_t hbase = (size_t)h * DHEAD * SEQ;
  const float scale = 0.125f;            // 64^-0.5

  // stage Q transposed: Qs[sq][d]  (global is [d][s], contiguous in s)
  for (int c = tid; c < 64 * 16; c += 256) {
    int d = c >> 4, sc = (c & 15) * 8;
    union { uint4 u; _Float16 hh[8]; } t;
    t.u = *(const uint4*)(Q + hbase + (size_t)d * SEQ + q0 + sc);
#pragma unroll
    for (int j = 0; j < 8; ++j) Qs[sc + j][d] = t.hh[j];
  }
  __syncthreads();

  // per-wave Q fragments (M=16 queries, K=d in two 32-steps) stay resident
  v16h qa0 = frag_a(&Qs[wid * 16][0],  72);
  v16h qa1 = frag_a(&Qs[wid * 16][32], 72);

  float mrun[8], lrun[8];
#pragma unroll
  for (int r = 0; r < 8; ++r) { mrun[r] = -1e30f; lrun[r] = 0.f; }
  v8f oacc[4];
#pragma unroll
  for (int nt = 0; nt < 4; ++nt) oacc[nt] = zero8();

  for (int j0 = 0; j0 < SEQ; j0 += 64) {
    // K block direct: Ks[d][sk]
    for (int c = tid; c < 64 * 8; c += 256) {
      int d = c >> 3, sc = (c & 7) * 8;
      *(uint4*)(&Ks[d][sc]) =
          *(const uint4*)(Km + hbase + (size_t)d * SEQ + j0 + sc);
    }
    // V block transposed: Vs[sk][d]
    for (int c = tid; c < 64 * 8; c += 256) {
      int d = c >> 3, sc = (c & 7) * 8;
      union { uint4 u; _Float16 hh[8]; } t;
      t.u = *(const uint4*)(V + hbase + (size_t)d * SEQ + j0 + sc);
#pragma unroll
      for (int j = 0; j < 8; ++j) Vs[sc + j][d] = t.hh[j];
    }
    __syncthreads();

    // S = Q^T K : M=16 queries, N=64 keys (4 tiles), K=d=64 (2 steps)
    v8f sacc[4];
#pragma unroll
    for (int nt = 0; nt < 4; ++nt) {
      sacc[nt] = zero8();
      v16h b0 = frag_b(&Ks[0][nt * 16],  72);
      sacc[nt] = wmma32(qa0, b0, sacc[nt]);
      v16h b1 = frag_b(&Ks[32][nt * 16], 72);
      sacc[nt] = wmma32(qa1, b1, sacc[nt]);
    }

    // ---- online softmax (rows live across the 16 lanes of each half) ----
    // 1) block row-max, reduced across the 16-lane group
#pragma unroll
    for (int r = 0; r < 8; ++r) {
      float v0 = fmaxf(fmaxf(sacc[0][r], sacc[1][r]),
                       fmaxf(sacc[2][r], sacc[3][r])) * scale;
#pragma unroll
      for (int off = 1; off < 16; off <<= 1)
        v0 = fmaxf(v0, __shfl_xor(v0, off, 32));
      // 2) fold into running max; rescale running sum + O immediately so the
      //    correction factor dies right here (short live range, no spills)
      float mnew = fmaxf(mrun[r], v0);
      float alpha = __expf(mrun[r] - mnew);
      mrun[r] = mnew;
      lrun[r] *= alpha;
#pragma unroll
      for (int nt = 0; nt < 4; ++nt) oacc[nt][r] *= alpha;
    }
    // 3) exponentiate, accumulate row sums, stage P in A-fragment layout
    float psum[8];
#pragma unroll
    for (int r = 0; r < 8; ++r) psum[r] = 0.f;
#pragma unroll
    for (int nt = 0; nt < 4; ++nt)
#pragma unroll
      for (int r = 0; r < 8; ++r) {
        float p = __expf(sacc[nt][r] * scale - mrun[r]);
        psum[r] += p;
        Ps[wid][r + hi * 8][nt * 16 + nl] = (_Float16)p;
      }
#pragma unroll
    for (int off = 1; off < 16; off <<= 1)
#pragma unroll
      for (int r = 0; r < 8; ++r)
        psum[r] += __shfl_xor(psum[r], off, 32);
#pragma unroll
    for (int r = 0; r < 8; ++r) lrun[r] += psum[r];

    // O += P V^T : M=16 queries, N=64 d (4 tiles), K=64 keys (2 steps)
#pragma unroll
    for (int ks = 0; ks < 2; ++ks) {
      v16h pa = frag_a(&Ps[wid][0][ks * 32], 72);
#pragma unroll
      for (int nt = 0; nt < 4; ++nt) {
        v16h bv = frag_b(&Vs[ks * 32][nt * 16], 72);
        oacc[nt] = wmma32(pa, bv, oacc[nt]);
      }
    }
    __syncthreads();   // protect Ks/Vs before next block overwrites them
  }

  // normalize and scatter O[h*64+d][sq]
#pragma unroll
  for (int nt = 0; nt < 4; ++nt) {
#pragma unroll
    for (int r = 0; r < 8; ++r) {
      int d  = nt * 16 + nl;
      int sq = q0 + wid * 16 + r + hi * 8;
      O[hbase + (size_t)d * SEQ + sq] = (_Float16)(oacc[nt][r] / lrun[r]);
    }
  }
}

// ---------------------------------------------------------------------------
// Host launch. Workspace (f16): x(2M) | Wq,Wk,Wv,Wo(1M each, contiguous so
// QKV is one stacked 3072x1024 GEMM) | q,k,v(2M each, contiguous) | attn(2M)
// = 14M halfs = 28 MB.
// ---------------------------------------------------------------------------
extern "C" void kernel_launch(void* const* d_in, const int* in_sizes, int n_in,
                              void* d_out, int out_size, void* d_ws, size_t ws_size,
                              hipStream_t stream) {
  const float* x  = (const float*)d_in[0];
  const float* Wq = (const float*)d_in[1];
  const float* Wk = (const float*)d_in[2];
  const float* Wv = (const float*)d_in[3];
  const float* Wo = (const float*)d_in[4];
  const float* bo = (const float*)d_in[5];
  float* out = (float*)d_out;

  const size_t MC = (size_t)CDIM * SEQ;    // 2M elements
  const size_t WC = (size_t)CDIM * CDIM;   // 1M elements
  _Float16* ws  = (_Float16*)d_ws;
  _Float16* xh  = ws;
  _Float16* wqh = xh  + MC;    // wq|wk|wv contiguous -> stacked [3072][1024]
  _Float16* wkh = wqh + WC;
  _Float16* wvh = wkh + WC;
  _Float16* woh = wvh + WC;
  _Float16* qh  = woh + WC;    // q|k|v contiguous -> stacked [3072][2048]
  _Float16* kh  = qh  + MC;
  _Float16* vh  = kh  + MC;
  _Float16* ah  = vh  + MC;

  // 1) fp32 -> f16
  int nx = (int)MC, nw = (int)WC;
  cvt_f32_f16<<<(nx + 255) / 256, 256, 0, stream>>>(x,  xh,  nx);
  cvt_f32_f16<<<(nw + 255) / 256, 256, 0, stream>>>(Wq, wqh, nw);
  cvt_f32_f16<<<(nw + 255) / 256, 256, 0, stream>>>(Wk, wkh, nw);
  cvt_f32_f16<<<(nw + 255) / 256, 256, 0, stream>>>(Wv, wvh, nw);
  cvt_f32_f16<<<(nw + 255) / 256, 256, 0, stream>>>(Wo, woh, nw);

  // 2) fused QKV projection: [3072][1024] x [1024][2048] -> [3072][2048] f16
  {
    dim3 grid(SEQ / 128, (3 * CDIM) / 128);   // (16, 24) = 384 WGs
    gemm_wmma<false><<<grid, 256, 0, stream>>>(
        wqh, xh, qh, nullptr, nullptr, 3 * CDIM, SEQ, CDIM);
  }

  // 3) flash attention per (head, 128-query block): (16, 16) = 256 WGs
  {
    dim3 grid(SEQ / 128, HEADS);
    attn_flash_wmma<<<grid, 256, 0, stream>>>(qh, kh, vh, ah);
  }

  // 4) output projection + bias, fp32 out
  {
    dim3 grid(SEQ / 128, CDIM / 128);         // (16, 8)
    gemm_wmma<true><<<grid, 256, 0, stream>>>(
        woh, ah, nullptr, out, bo, CDIM, SEQ, CDIM);
  }
}
